// CAM_3289944949057
// MI455X (gfx1250) — matrix operand order repaired and checked by reading, not compiled, not run on hardware
//
#include <hip/hip_runtime.h>
#include <hip/hip_bf16.h>

typedef __attribute__((ext_vector_type(8)))  __bf16 v8bf;
typedef __attribute__((ext_vector_type(16))) __bf16 v16bf;
typedef __attribute__((ext_vector_type(8)))  float  v8f;

#define WMMA_BF16(A, B, C) \
    __builtin_amdgcn_wmma_f32_16x16x32_bf16(false, (A), false, (B), (short)0, (C), false, false)

static __device__ __forceinline__ v16bf cat8(v8bf a, v8bf b) {
    return __builtin_shufflevector(a, b, 0,1,2,3,4,5,6,7,8,9,10,11,12,13,14,15);
}

// A operand (16x32 bf16): lane holds row M=lane&15; element e -> k = 16*half + e (contiguous 16).
static __device__ __forceinline__ v16bf load_a_bf(const __bf16* __restrict__ row, int half) {
    v8bf c0 = *reinterpret_cast<const v8bf*>(row + 16 * half);
    v8bf c1 = *reinterpret_cast<const v8bf*>(row + 16 * half + 8);
    return cat8(c0, c1);
}
// B operand (32x16 bf16), source rows are the B-columns: element e -> k = 16*half + e (contiguous 16).
static __device__ __forceinline__ v16bf load_b_bf(const __bf16* __restrict__ row, int half) {
    v8bf c0 = *reinterpret_cast<const v8bf*>(row + 16 * half);
    v8bf c1 = *reinterpret_cast<const v8bf*>(row + 16 * half + 8);
    return cat8(c0, c1);
}

// ---------------- Kernel 0: split x (f32) into bf16 hi + residual lo, once ----------------
__global__ void cam_split(const float* __restrict__ x,
                          __bf16* __restrict__ xh, __bf16* __restrict__ xl) {
    const size_t i = ((size_t)blockIdx.x * blockDim.x + threadIdx.x) * 8;
    float4 q0 = *reinterpret_cast<const float4*>(x + i);
    float4 q1 = *reinterpret_cast<const float4*>(x + i + 4);
    float v[8] = {q0.x, q0.y, q0.z, q0.w, q1.x, q1.y, q1.z, q1.w};
    v8bf h, l;
#pragma unroll
    for (int j = 0; j < 8; ++j) {
        __bf16 hh = (__bf16)v[j];
        h[j] = hh;
        l[j] = (__bf16)(v[j] - (float)hh);
    }
    *reinterpret_cast<v8bf*>(xh + i) = h;
    *reinterpret_cast<v8bf*>(xl + i) = l;
}

// ---------------- Kernel 1: S[b] = X[b] * X[b]^T  (512x512, K=4096) ----------------
// One wave computes a 32x32 block of S: 16 b128 loads + 12 WMMAs per K=32 step.
__global__ void cam_gram(const __bf16* __restrict__ xh, const __bf16* __restrict__ xl,
                         float* __restrict__ S) {
    const int lane = threadIdx.x;          // one wave per block
    const int half = lane >> 4;
    const int l15  = lane & 15;
    const int ti = blockIdx.x;             // 0..15 (32-row block)
    const int tj = blockIdx.y;             // 0..15 (32-col block)
    const int b  = blockIdx.z;             // 0..15

    const size_t a0off = ((size_t)b * 512 + (size_t)ti * 32 + l15) * 4096;
    const size_t b0off = ((size_t)b * 512 + (size_t)tj * 32 + l15) * 4096;
    const __bf16* a0h_row = xh + a0off;
    const __bf16* a0l_row = xl + a0off;
    const __bf16* b0h_row = xh + b0off;
    const __bf16* b0l_row = xl + b0off;
    const size_t rstep = (size_t)16 * 4096;

    v8f c00 = {}, c01 = {}, c10 = {}, c11 = {};
    for (int k = 0; k < 4096; k += 32) {
        v16bf a0h = load_a_bf(a0h_row + k, half);
        v16bf a0l = load_a_bf(a0l_row + k, half);
        v16bf a1h = load_a_bf(a0h_row + rstep + k, half);
        v16bf a1l = load_a_bf(a0l_row + rstep + k, half);
        v16bf b0h = load_b_bf(b0h_row + k, half);
        v16bf b0l = load_b_bf(b0l_row + k, half);
        v16bf b1h = load_b_bf(b0h_row + rstep + k, half);
        v16bf b1l = load_b_bf(b0l_row + rstep + k, half);
        // Round-robin across 4 independent accumulators to hide WMMA latency.
        c00 = WMMA_BF16(a0h, b0h, c00);
        c01 = WMMA_BF16(a0h, b1h, c01);
        c10 = WMMA_BF16(a1h, b0h, c10);
        c11 = WMMA_BF16(a1h, b1h, c11);
        c00 = WMMA_BF16(a0h, b0l, c00);
        c01 = WMMA_BF16(a0h, b1l, c01);
        c10 = WMMA_BF16(a1h, b0l, c10);
        c11 = WMMA_BF16(a1h, b1l, c11);
        c00 = WMMA_BF16(a0l, b0h, c00);
        c01 = WMMA_BF16(a0l, b1h, c01);
        c10 = WMMA_BF16(a1l, b0h, c10);
        c11 = WMMA_BF16(a1l, b1h, c11);
    }

    float* s00 = S + ((size_t)b * 512 + (size_t)ti * 32) * 512 + (size_t)tj * 32;
#pragma unroll
    for (int r = 0; r < 8; ++r) {
        const size_t ro = (size_t)(r + 8 * half) * 512;
        s00[ro + l15]                         = c00[r];
        s00[ro + 16 + l15]                    = c01[r];
        s00[ro + (size_t)16 * 512 + l15]      = c10[r];
        s00[ro + (size_t)16 * 512 + 16 + l15] = c11[r];
    }
}

// ---------------- Kernel 2: row softmax over 512 logits -> bf16 hi/lo split P ----------------
__global__ void cam_softmax(const float* __restrict__ S,
                            __bf16* __restrict__ Ph, __bf16* __restrict__ Pl) {
    __shared__ float red[256];
    const int t = threadIdx.x;                        // 256 threads
    const size_t row = (size_t)blockIdx.x * 512;      // one row per block
    const float* p = S + row;

    float a0 = p[t], a1 = p[t + 256];
    red[t] = fmaxf(a0, a1);
    __syncthreads();
    for (int s = 128; s > 0; s >>= 1) {
        if (t < s) red[t] = fmaxf(red[t], red[t + s]);
        __syncthreads();
    }
    const float rmax = red[0];
    __syncthreads();

    float e0 = __expf(a0 - rmax), e1 = __expf(a1 - rmax);
    red[t] = e0 + e1;
    __syncthreads();
    for (int s = 128; s > 0; s >>= 1) {
        if (t < s) red[t] += red[t + s];
        __syncthreads();
    }
    const float inv = 1.0f / red[0];

    float p0 = e0 * inv, p1 = e1 * inv;
    __bf16 h0 = (__bf16)p0, h1 = (__bf16)p1;
    Ph[row + t]       = h0;
    Pl[row + t]       = (__bf16)(p0 - (float)h0);
    Ph[row + t + 256] = h1;
    Pl[row + t + 256] = (__bf16)(p1 - (float)h1);
}

// ---------------- Kernel 3: out = beta * (P[b] @ X[b]) + x  (M=512, K=512, N=4096) ----------------
// One wave computes a 32x32 block of Y (2x2 tiles): 12 WMMAs per K=32 step.
__global__ void cam_av(const float* __restrict__ x,
                       const __bf16* __restrict__ Ph, const __bf16* __restrict__ Pl,
                       const __bf16* __restrict__ xh, const __bf16* __restrict__ xl,
                       const float* __restrict__ beta, float* __restrict__ out) {
    const int lane = threadIdx.x;          // one wave per block
    const int half = lane >> 4;
    const int l15  = lane & 15;
    const int tn = blockIdx.x;             // 0..127 (32-col block)
    const int tm = blockIdx.y;             // 0..15  (32-row block)
    const int b  = blockIdx.z;

    const size_t aoff = ((size_t)b * 512 + (size_t)tm * 32 + l15) * 512;
    const __bf16* a0h_row = Ph + aoff;
    const __bf16* a0l_row = Pl + aoff;
    const size_t rstep = (size_t)16 * 512;
    const size_t xboff = (size_t)b * 512 * 4096;
    const int col0 = tn * 32 + l15;        // lane's first B/D column (col1 = col0 + 16)

    v8f c00 = {}, c01 = {}, c10 = {}, c11 = {};
    for (int k = 0; k < 512; k += 32) {
        v16bf a0h = load_a_bf(a0h_row + k, half);
        v16bf a0l = load_a_bf(a0l_row + k, half);
        v16bf a1h = load_a_bf(a0h_row + rstep + k, half);
        v16bf a1l = load_a_bf(a0l_row + rstep + k, half);
        // B[k][n] = X[k][col]: element e -> k = kbase + 16*half + e, column fixed per lane.
        v16bf b0h, b0l, b1h, b1l;
        const size_t bofs = xboff + (size_t)(k + 16 * half) * 4096 + col0;
#pragma unroll
        for (int e = 0; e < 16; ++e) {
            const size_t o = bofs + (size_t)e * 4096;
            b0h[e] = xh[o];
            b0l[e] = xl[o];
            b1h[e] = xh[o + 16];
            b1l[e] = xl[o + 16];
        }
        c00 = WMMA_BF16(a0h, b0h, c00);
        c01 = WMMA_BF16(a0h, b1h, c01);
        c10 = WMMA_BF16(a1h, b0h, c10);
        c11 = WMMA_BF16(a1h, b1h, c11);
        c00 = WMMA_BF16(a0h, b0l, c00);
        c01 = WMMA_BF16(a0h, b1l, c01);
        c10 = WMMA_BF16(a1h, b0l, c10);
        c11 = WMMA_BF16(a1h, b1l, c11);
        c00 = WMMA_BF16(a0l, b0h, c00);
        c01 = WMMA_BF16(a0l, b1h, c01);
        c10 = WMMA_BF16(a1l, b0h, c10);
        c11 = WMMA_BF16(a1l, b1h, c11);
    }

    const float bv = beta[0];
#pragma unroll
    for (int r = 0; r < 8; ++r) {
        const int row0 = tm * 32 + r + 8 * half;
        size_t i00 = ((size_t)b * 512 + row0) * 4096 + col0;
        size_t i10 = i00 + (size_t)16 * 4096;
        out[i00]      = bv * c00[r] + x[i00];
        out[i00 + 16] = bv * c01[r] + x[i00 + 16];
        out[i10]      = bv * c10[r] + x[i10];
        out[i10 + 16] = bv * c11[r] + x[i10 + 16];
    }
}

extern "C" void kernel_launch(void* const* d_in, const int* in_sizes, int n_in,
                              void* d_out, int out_size, void* d_ws, size_t ws_size,
                              hipStream_t stream) {
    const float* x    = (const float*)d_in[0];   // [16,512,64,64] f32 (134 MB)
    const float* beta = (const float*)d_in[1];   // [1] f32
    float* out = (float*)d_out;                  // [16,512,64,64] f32

    // Workspace layout (~168 MB total):
    //   S   : 16*512*512 f32            = 16,777,216 B
    //   Ph  : 16*512*512 bf16           =  8,388,608 B
    //   Pl  : 16*512*512 bf16           =  8,388,608 B
    //   Xh  : 16*512*4096 bf16          = 67,108,864 B
    //   Xl  : 16*512*4096 bf16          = 67,108,864 B
    char* ws = (char*)d_ws;
    float*  S  = (float*)ws;
    __bf16* Ph = (__bf16*)(ws + 16777216);
    __bf16* Pl = Ph + (size_t)16 * 512 * 512;
    __bf16* Xh = (__bf16*)(ws + 16777216 + 2 * 8388608);
    __bf16* Xl = Xh + (size_t)16 * 512 * 4096;

    // Phase 0: one coalesced pass splitting x into bf16 hi/lo.
    const size_t nelem = (size_t)16 * 512 * 4096;
    cam_split<<<dim3((unsigned)(nelem / (256 * 8))), dim3(256), 0, stream>>>(x, Xh, Xl);

    // Phase 1: gram logits.
    cam_gram<<<dim3(16, 16, 16), dim3(32), 0, stream>>>(Xh, Xl, S);

    // Phase 2: softmax + split P.
    cam_softmax<<<dim3(16 * 512), dim3(256), 0, stream>>>(S, Ph, Pl);

    // Phase 3: attention-weighted values + epilogue.
    cam_av<<<dim3(128, 16, 16), dim3(32), 0, stream>>>(x, Ph, Pl, Xh, Xl, beta, out);
}
